// BertSelfAttention_2851858284546
// MI455X (gfx1250) — matrix operand order repaired and checked by reading, not compiled, not run on hardware
//
#include <hip/hip_runtime.h>

// ---------------------------------------------------------------------------
// CDNA5 (gfx1250) fused BERT self-attention forward.
// bf16 WMMA (v_wmma_f32_16x16x32_bf16), f32 accumulation, flash-attention
// online softmax with post-softmax C_prior fused in.
// All WMMA fragment loads are 128-bit; tile staging uses the gfx1250 async
// global->LDS path (ASYNCcnt) with double buffering when available.
// Shapes hardcoded to the reference: B=2, S=2048, D=1024, H=16, W=64.
// ---------------------------------------------------------------------------

typedef __bf16 bf16_t;
typedef __attribute__((ext_vector_type(16))) bf16_t         v16bf;
typedef __attribute__((ext_vector_type(8)))  float          v8f;
typedef __attribute__((ext_vector_type(8)))  unsigned short u16x8;  // 16 B
typedef __attribute__((ext_vector_type(4)))  int            v4i;    // 16 B

#define S_LEN   2048
#define D_DIM   1024
#define H_NUM   16
#define W_DIM   64
#define NEG_BIG (-1.0e15f)   // matches reference NEG_INF fill

// --------------------------- async global->LDS -----------------------------
#if __has_builtin(__builtin_amdgcn_global_load_async_to_lds_b128)
#define HAVE_ASYNC_LDS 1
#else
#define HAVE_ASYNC_LDS 0
#endif

// pointer-to-v4i in global (AS1) / LDS (AS3) address spaces
typedef __attribute__((address_space(1))) v4i* gv4i_p;
typedef __attribute__((address_space(3))) v4i* lv4i_p;

// copy 16 bytes global -> LDS (async when the toolchain exposes it)
__device__ __forceinline__ void cp16(unsigned short* lds_dst,
                                     const unsigned short* gsrc) {
#if HAVE_ASYNC_LDS
  __builtin_amdgcn_global_load_async_to_lds_b128(
      (gv4i_p)(uintptr_t)(gsrc),
      (lv4i_p)(unsigned)(uintptr_t)(lds_dst), 0, 0);
#else
  *(u16x8*)lds_dst = *(const u16x8*)gsrc;
#endif
}

#if HAVE_ASYNC_LDS
#if __has_builtin(__builtin_amdgcn_s_wait_asynccnt)
#define ASYNC_WAIT(n) __builtin_amdgcn_s_wait_asynccnt(n)
#else
#define ASYNC_WAIT(n) asm volatile("s_wait_asynccnt %0" ::"n"(n) : "memory")
#endif
#else
#define ASYNC_WAIT(n) do { } while (0)
#endif

// ---------------------------------------------------------------------------
// float -> bf16 bits, round-to-nearest-even
__device__ __forceinline__ unsigned short f2bf_bits(float f) {
  union { float f; unsigned u; } in; in.f = f;
  unsigned u = in.u;
  u += 0x7FFFu + ((u >> 16) & 1u);
  return (unsigned short)(u >> 16);
}

// Assemble a v16bf A/B fragment from two contiguous 16-byte runs.
union frag_u {
  u16x8 h[2];
  v16bf v;
};
__device__ __forceinline__ v16bf load_frag(const unsigned short* p0,
                                           const unsigned short* p1) {
  frag_u f;
  f.h[0] = *(const u16x8*)p0;
  f.h[1] = *(const u16x8*)p1;
  return f.v;
}

__device__ __forceinline__ v8f wmma_bf16(v16bf a, v16bf b, v8f c) {
  return __builtin_amdgcn_wmma_f32_16x16x32_bf16(
      /*neg_a=*/false, a, /*neg_b=*/false, b,
      /*c_mod=*/(short)0, c, /*reuse_a=*/false, /*reuse_b=*/false);
}

// A-fragment K permutation: LDS slot within a row holds K:
//   k in [0,8)->slot k, [8,16)->k+8, [16,24)->k-8, [24,32)->k
// i.e. 8-wide block order {0,2,1,3}. A-frag read = slots [16*khalf, +16).
__device__ __forceinline__ int ablk(int blk) {
  const int perm[4] = {0, 2, 1, 3};
  return perm[blk];
}

// ---------------------------------------------------------------------------
// fp32 -> bf16 conversion (stored as raw bits)
// ---------------------------------------------------------------------------
__global__ void cvt_f32_bf16(const float* __restrict__ in,
                             unsigned short* __restrict__ o, int n) {
  int i = blockIdx.x * blockDim.x + threadIdx.x;
  if (i < n) o[i] = f2bf_bits(in[i]);
}

// ---------------------------------------------------------------------------
// QKV projection: out = x @ W^T + b, written bf16 in (B,H,S,W) layout.
// blockIdx.z selects q/k/v. Block = 256 threads = 8 waves.
// Tile: BM=128, BN=64, BK=32; wave owns a 32x32 C tile (4 WMMA accums,
// 2 A-frags x 2 B-frags -> 4 WMMAs per 8 b128 fragment loads).
// Double-buffered LDS, async staging.
// LDS: As [m][32 K, A-permuted] stride 40 halves; Ws [n][32 K, natural]
// (W is already N x K row-major and B-frag K is contiguous -> straight copy).
// ---------------------------------------------------------------------------
__global__ __launch_bounds__(256) void qkv_gemm(
    const unsigned short* __restrict__ xbf,
    const unsigned short* __restrict__ wqm, const unsigned short* __restrict__ wkm,
    const unsigned short* __restrict__ wvm,
    const float* __restrict__ bq, const float* __restrict__ bk,
    const float* __restrict__ bv,
    unsigned short* __restrict__ qo, unsigned short* __restrict__ ko,
    unsigned short* __restrict__ vo) {
  __shared__ unsigned short As[2][128 * 40];
  __shared__ unsigned short Ws[2][64 * 40];

  const int t = threadIdx.x;
  const int z = blockIdx.z;
  const unsigned short* Wm = (z == 0) ? wqm : (z == 1) ? wkm : wvm;
  const float* bias        = (z == 0) ? bq  : (z == 1) ? bk  : bv;
  unsigned short* dst      = (z == 0) ? qo  : (z == 1) ? ko  : vo;

  const int m0 = blockIdx.y * 128;
  const int n0 = blockIdx.x * 64;
  const int lane = t & 31, wid = t >> 5;
  const int lm = lane & 15, khalf = lane >> 4;
  const int wm = (wid & 3) * 32;        // 4 m-positions x 32 rows
  const int wn = (wid >> 2) * 32;       // 2 n-positions x 32 cols

  // staging indices: A 128x32 = 2 x 16B per thread; W 64x32 = 1 x 16B
  const int ar = t >> 1, ap = (t & 1) * 2;        // A row, first src block
  const int wr = t >> 2, wb = t & 3;              // W row, src block
  const unsigned short* gA = xbf + (size_t)(m0 + ar) * D_DIM + ap * 8;
  const unsigned short* gW = Wm  + (size_t)(n0 + wr) * D_DIM + wb * 8;
  const int adst0 = ar * 40 + ablk(ap) * 8;
  const int adst1 = ar * 40 + ablk(ap + 1) * 8;
  const int wdst  = wr * 40 + wb * 8;

  v8f acc[2][2] = {};

#define STAGE(buf, k0)                                  \
  do {                                                  \
    cp16(&As[buf][adst0], gA + (k0));                   \
    cp16(&As[buf][adst1], gA + (k0) + 8);               \
    cp16(&Ws[buf][wdst],  gW + (k0));                   \
  } while (0)

  STAGE(0, 0);
  const int KT = D_DIM / 32;
  for (int kt = 0; kt < KT; ++kt) {
    const int cur = kt & 1;
    if (kt + 1 < KT) {
      STAGE(1 - cur, (kt + 1) * 32);   // prefetch next tile (buffer freed by
      ASYNC_WAIT(3);                   //  the barrier at end of prev iter)
    } else {
      ASYNC_WAIT(0);
    }
    __syncthreads();

    const unsigned short* a0p = &As[cur][(wm + lm) * 40 + khalf * 16];
    const unsigned short* a1p = a0p + 16 * 40;
    const unsigned short* b0p = &Ws[cur][(wn + lm) * 40 + khalf * 8];
    const unsigned short* b1p = b0p + 16 * 40;
    v16bf a0 = load_frag(a0p, a0p + 8);
    v16bf a1 = load_frag(a1p, a1p + 8);
    v16bf b0 = load_frag(b0p, b0p + 16);
    v16bf b1 = load_frag(b1p, b1p + 16);
    acc[0][0] = wmma_bf16(a0, b0, acc[0][0]);
    acc[0][1] = wmma_bf16(a0, b1, acc[0][1]);
    acc[1][0] = wmma_bf16(a1, b0, acc[1][0]);
    acc[1][1] = wmma_bf16(a1, b1, acc[1][1]);
    __syncthreads();
  }
#undef STAGE

#pragma unroll
  for (int i = 0; i < 2; ++i)
#pragma unroll
    for (int j = 0; j < 2; ++j)
#pragma unroll
      for (int r = 0; r < 8; ++r) {
        int m = m0 + wm + i * 16 + r + 8 * khalf;   // row in (B*S)
        int n = n0 + wn + j * 16 + lm;              // col in D
        float val = acc[i][j][r] + bias[n];
        int bb = m >> 11, ss = m & (S_LEN - 1);
        int hh = n >> 6,  ww = n & (W_DIM - 1);
        dst[(((size_t)(bb * H_NUM + hh)) * S_LEN + ss) * W_DIM + ww] =
            f2bf_bits(val);
      }
}

// ---------------------------------------------------------------------------
// Flash attention with fused post-softmax C_prior.
// grid = (S/128, B*H); block = 256 = 8 waves; wave owns 16 q rows.
// LDS: Kt [kcol][w]   stride 72 halves (score B-frags read contiguous w)
//      VtT [w][kcol]  stride 40 halves (PV B-frags read contiguous kcol)
//      Pt  per-wave [m][32, A-permuted] stride 40 halves
// ---------------------------------------------------------------------------
__global__ __launch_bounds__(256) void attn_kernel(
    const unsigned short* __restrict__ qbf, const unsigned short* __restrict__ kbf,
    const unsigned short* __restrict__ vbf, const int* __restrict__ amask,
    const float* __restrict__ Cp, float* __restrict__ out) {
  __shared__ unsigned short Kt[32 * 72];
  __shared__ unsigned short VtT[64 * 40];
  __shared__ unsigned short Pt[8][16 * 40];

  const int t = threadIdx.x;
  const int lane = t & 31, wid = t >> 5;
  const int lm = lane & 15, khalf = lane >> 4;
  const int bh = blockIdx.y;
  const int b = bh >> 4, h = bh & 15;
  const int q0 = blockIdx.x * 128 + wid * 16;

  // q fragments: 16 rows x 64 K as two 16x32 A-frags, direct 128-bit loads
  // (A-frag lane stream = K run [16k, 16k+16), k=khalf)
  v16bf qA[2];
  const unsigned short* qrowp = qbf + ((size_t)bh * S_LEN + (q0 + lm)) * W_DIM;
#pragma unroll
  for (int c = 0; c < 2; ++c)
    qA[c] = load_frag(qrowp + c * 32 + 16 * khalf,
                      qrowp + c * 32 + 8 + 16 * khalf);

  v8f acc[4] = {};
  float mrow[8], Zr[8];
#pragma unroll
  for (int r = 0; r < 8; ++r) { mrow[r] = -INFINITY; Zr[r] = 0.f; }

  const int ldr = t >> 3, ldc = (t & 7) * 8;  // 256*8 = 2048 = 32x64
  // P*C staging slots (A-permutation of kcol=lm and kcol=16+lm)
  const int ppos0 = (lm < 8) ? lm : (lm + 8);
  const int ppos1 = ppos0 + 8;

  for (int kt = 0; kt < S_LEN / 32; ++kt) {
    const int k0 = kt * 32;
    const unsigned short* gk = kbf + ((size_t)bh * S_LEN + k0 + ldr) * W_DIM + ldc;
    const unsigned short* gv = vbf + ((size_t)bh * S_LEN + k0 + ldr) * W_DIM + ldc;
    cp16(&Kt[ldr * 72 + ldc], gk);                      // [kcol][w], async path
    u16x8 vv = *(const u16x8*)gv;                       // transpose V -> [w][kcol]
#pragma unroll
    for (int j = 0; j < 8; ++j) VtT[(ldc + j) * 40 + ldr] = vv[j];
    ASYNC_WAIT(0);
    __syncthreads();

    // scores = q @ k^T : two 16x16 tiles (cols k0+lm, k0+16+lm)
    v8f s0 = {}, s1 = {};
#pragma unroll
    for (int wc = 0; wc < 2; ++wc) {
      const unsigned short* kp0 = &Kt[lm * 72 + wc * 32 + khalf * 16];
      const unsigned short* kp1 = &Kt[(lm + 16) * 72 + wc * 32 + khalf * 16];
      v16bf kb0 = load_frag(kp0, kp0 + 8);
      v16bf kb1 = load_frag(kp1, kp1 + 8);
      s0 = wmma_bf16(qA[wc], kb0, s0);
      s1 = wmma_bf16(qA[wc], kb1, s1);
    }

    const int msk0 = amask[b * S_LEN + k0 + lm];
    const int msk1 = amask[b * S_LEN + k0 + 16 + lm];
    float tm[8];
#pragma unroll
    for (int r = 0; r < 8; ++r) {
      float a0 = s0[r] * 0.125f; if (msk0 == 0) a0 = NEG_BIG;
      float a1 = s1[r] * 0.125f; if (msk1 == 0) a1 = NEG_BIG;
      s0[r] = a0; s1[r] = a1;
      tm[r] = fmaxf(a0, a1);
    }
    // row max over the 16 lanes of this khalf group
#pragma unroll
    for (int r = 0; r < 8; ++r)
#pragma unroll
      for (int xm = 1; xm < 16; xm <<= 1)
        tm[r] = fmaxf(tm[r], __shfl_xor(tm[r], xm, 32));

#pragma unroll
    for (int r = 0; r < 8; ++r) {
      float mn   = fmaxf(mrow[r], tm[r]);
      float corr = __expf(mrow[r] - mn);     // exp(-inf)=0 first time
      mrow[r] = mn;
      float p0 = __expf(s0[r] - mn);
      float p1 = __expf(s1[r] - mn);
      s0[r] = p0; s1[r] = p1;
      float rs = p0 + p1;
#pragma unroll
      for (int xm = 1; xm < 16; xm <<= 1) rs += __shfl_xor(rs, xm, 32);
      Zr[r] = Zr[r] * corr + rs;             // softmax denom (no C_prior)
#pragma unroll
      for (int c = 0; c < 4; ++c) acc[c][r] *= corr;
    }

    // P*C_prior staged directly into A-fragment (permuted) order
#pragma unroll
    for (int r = 0; r < 8; ++r) {
      int qrow = q0 + r + 8 * khalf;
      const float* crow = Cp + ((size_t)b * S_LEN + qrow) * S_LEN + k0;
      unsigned short* prow = &Pt[wid][(r + 8 * khalf) * 40];
      prow[ppos0] = f2bf_bits(s0[r] * crow[lm]);
      prow[ppos1] = f2bf_bits(s1[r] * crow[16 + lm]);
    }
    asm volatile("s_wait_dscnt 0x0" ::: "memory");  // wave-local LDS RAW

    const unsigned short* pp = &Pt[wid][lm * 40 + khalf * 16];
    v16bf pA = load_frag(pp, pp + 8);

    // acc += P@V : four 16x16 output chunks over W=64, K=32
#pragma unroll
    for (int wc = 0; wc < 4; ++wc) {
      const unsigned short* vp = &VtT[(wc * 16 + lm) * 40 + khalf * 8];
      v16bf vb = load_frag(vp, vp + 16);
      acc[wc] = wmma_bf16(pA, vb, acc[wc]);
    }
    __syncthreads();
  }

#pragma unroll
  for (int wc = 0; wc < 4; ++wc)
#pragma unroll
    for (int r = 0; r < 8; ++r) {
      int qrow = q0 + r + 8 * khalf;
      int w = wc * 16 + lm;
      out[((size_t)b * S_LEN + qrow) * D_DIM + h * W_DIM + w] =
          acc[wc][r] / Zr[r];
    }
}

// ---------------------------------------------------------------------------
extern "C" void kernel_launch(void* const* d_in, const int* in_sizes, int n_in,
                              void* d_out, int out_size, void* d_ws,
                              size_t ws_size, hipStream_t stream) {
  const float* x   = (const float*)d_in[0];
  const int*   am  = (const int*)d_in[1];
  const float* Cp  = (const float*)d_in[2];
  const float* Wq  = (const float*)d_in[3];
  const float* bq  = (const float*)d_in[4];
  const float* Wk  = (const float*)d_in[5];
  const float* bk  = (const float*)d_in[6];
  const float* Wv  = (const float*)d_in[7];
  const float* bv  = (const float*)d_in[8];
  float* out = (float*)d_out;

  // workspace layout (bf16 bits): x(4096x1024), 3x W(1024x1024), q/k/v
  char* ws = (char*)d_ws;
  const size_t NX = (size_t)4096 * 1024;   // B*S*D
  const size_t NW = (size_t)1024 * 1024;
  unsigned short* xbf  = (unsigned short*)ws;
  unsigned short* wqbf = (unsigned short*)(ws + NX * 2);
  unsigned short* wkbf = wqbf + NW;
  unsigned short* wvbf = wkbf + NW;
  unsigned short* qbf  = wvbf + NW;
  unsigned short* kbf  = qbf + NX;
  unsigned short* vbf  = kbf + NX;

  cvt_f32_bf16<<<(int)((NX + 255) / 256), 256, 0, stream>>>(x, xbf, (int)NX);
  cvt_f32_bf16<<<(int)((NW + 255) / 256), 256, 0, stream>>>(Wq, wqbf, (int)NW);
  cvt_f32_bf16<<<(int)((NW + 255) / 256), 256, 0, stream>>>(Wk, wkbf, (int)NW);
  cvt_f32_bf16<<<(int)((NW + 255) / 256), 256, 0, stream>>>(Wv, wvbf, (int)NW);

  // M=4096 rows x N=1024 cols, 128x64 tiles; z = q/k/v
  qkv_gemm<<<dim3(1024 / 64, 4096 / 128, 3), 256, 0, stream>>>(
      xbf, wqbf, wkbf, wvbf, bq, bk, bv, qbf, kbf, vbf);

  // (S/128) q-strips x (B*H) heads
  attn_kernel<<<dim3(S_LEN / 128, 32), 256, 0, stream>>>(
      qbf, kbf, vbf, am, Cp, out);
}
